// RUMLayer_39642548142869
// MI455X (gfx1250) — compile-verified, matrix operand order checked by reading, not run on hardware
//
#include <hip/hip_runtime.h>
#include <hip/hip_bf16.h>
#include <math.h>

typedef __attribute__((ext_vector_type(16))) _Float16 v16h;
typedef __attribute__((ext_vector_type(8)))  float    v8f;

#define SDIM   4
#define NNODE  5000
#define LWALK  8
#define INF_   128
#define OUTF   128
#define EFEAT  64
#define NEDGE  80000
#define DINF   385
#define BROWS  20000
#define GDIM   384
#define HP     136   // padded half-stride for 16x128 h tile (4-bank rotation)
#define XP     392   // padded half-stride for 16x385 x tile (784B = 49*16B)

__device__ __forceinline__ int koff(int p, int laneHi) {
  // 16-bit A/B fragment K-offset within a 32-wide K chunk (ISA 7.12.2)
  return (((p >> 1) & 3) << 1) + (p & 1) + ((p >> 3) << 4) + (laneHi << 3);
}

__device__ __forceinline__ v8f wmma_f16(v16h a, v16h b, v8f c) {
  return __builtin_amdgcn_wmma_f32_16x16x32_f16(false, a, false, b, (short)0, c,
                                                false, false);
}

// Branch-free activations: v_exp_f32 + v_rcp_f32 only (no EXEC manipulation,
// keeps the WMMA-pipelined loop straight-line as required by WMMA EXEC rules).
__device__ __forceinline__ float sigm(float x) {
  return __builtin_amdgcn_rcpf(1.0f + __expf(-x));
}
__device__ __forceinline__ float tanh_fast(float x) {
  return fmaf(2.0f, __builtin_amdgcn_rcpf(1.0f + __expf(-2.0f * x)), -1.0f);
}

// ---------------------------------------------------------------- prep -----
__global__ __launch_bounds__(256) void prep_kernel(const int* __restrict__ walks,
                                                   const int* __restrict__ deg,
                                                   float* __restrict__ usc,
                                                   int* __restrict__ maxdeg) {
  int b = blockIdx.x * blockDim.x + threadIdx.x;
  if (b >= BROWS) return;
  int wl[LWALK];
#pragma unroll
  for (int l = 0; l < LWALK; ++l) wl[l] = walks[b * LWALK + l];
  int dmax = 0;
#pragma unroll
  for (int l = 0; l < LWALK; ++l) {
    int j = 0;
    while (wl[j] != wl[l]) ++j;              // first occurrence (<= l)
    float u = (float)j * (6.283185307179586f / (float)LWALK);
    usc[((size_t)b * LWALK + (LWALK - 1 - l)) * 2 + 0] = sinf(u);
    usc[((size_t)b * LWALK + (LWALK - 1 - l)) * 2 + 1] = cosf(u);
    int d = deg[wl[l]];
    dmax = d > dmax ? d : dmax;
  }
  atomicMax(maxdeg, dmax);
}

// ------------------------------------------------------- M2 = Wih @ W_e ----
__global__ __launch_bounds__(256) void m2_kernel(const float* __restrict__ Wih,
                                                 const float* __restrict__ We,
                                                 float* __restrict__ M2) {
  int idx = blockIdx.x * blockDim.x + threadIdx.x;
  if (idx >= GDIM * EFEAT) return;
  int g = idx >> 6, c = idx & 63;
  float acc = 0.f;
  for (int d = 0; d < DINF; ++d) acc += Wih[(size_t)g * DINF + d] * We[(size_t)d * EFEAT + c];
  M2[(size_t)g * EFEAT + c] = acc;
}

// ----------------------------------------- pack weights into B-fragments ---
// W is [G=384][K] row-major; B-matrix is Kx G (B[k][g] = W[g][k]).
// Fragment (nt,kt): dst[((nt*KT+kt)*32+lane)*16+p] = W[nt*16+(lane&15)][kt*32+koff]
__global__ __launch_bounds__(256) void pack_kernel(
    const float* __restrict__ Whh_f, const float* __restrict__ Whh_b,
    const float* __restrict__ Whh,   const float* __restrict__ Wih,
    const float* __restrict__ M2,
    _Float16* pWhh_f, _Float16* pWhh_b, _Float16* pWhh,
    _Float16* pWihK, _Float16* pM2) {
  const float* src; int K, ldw; _Float16* dst;
  switch (blockIdx.y) {
    case 0: src = Whh_f; K = 128; ldw = 128; dst = pWhh_f; break;
    case 1: src = Whh_b; K = 128; ldw = 128; dst = pWhh_b; break;
    case 2: src = Whh;   K = 128; ldw = 128; dst = pWhh;   break;
    case 3: src = Wih;   K = 384; ldw = DINF; dst = pWihK; break;
    default: src = M2;   K = 64;  ldw = 64;  dst = pM2;    break;
  }
  int KT = K >> 5;
  int idx = blockIdx.x * blockDim.x + threadIdx.x;
  if (idx >= 24 * KT * 32) return;
  int lane = idx & 31;
  int kt = (idx >> 5) % KT;
  int nt = (idx >> 5) / KT;
  int g = nt * 16 + (lane & 15);
  int laneHi = (lane >> 4) & 1;
#pragma unroll
  for (int p = 0; p < 16; ++p) {
    int k = kt * 32 + koff(p, laneHi);
    dst[((size_t)(nt * KT + kt) * 32 + lane) * 16 + p] =
        (_Float16)src[(size_t)g * ldw + k];
  }
}

// ------------------------------------------------- bidirectional walk GRU --
__global__ __launch_bounds__(256) void walk_gru_kernel(
    const float* __restrict__ usc,
    const float* __restrict__ Wih_f, const float* __restrict__ bih_f,
    const float* __restrict__ bhh_f,
    const float* __restrict__ Wih_b, const float* __restrict__ bih_b,
    const float* __restrict__ bhh_b,
    const _Float16* __restrict__ pWhh_f, const _Float16* __restrict__ pWhh_b,
    _Float16* __restrict__ ywalk, _Float16* __restrict__ hdir) {
  const int dir = blockIdx.y;
  const float* Wd = dir ? Wih_b : Wih_f;
  const float* bi = dir ? bih_b : bih_f;
  const float* bh = dir ? bhh_b : bhh_f;
  const _Float16* pW = dir ? pWhh_b : pWhh_f;

  const int b0 = blockIdx.x * 16;
  const int tid = threadIdx.x;
  const int lane = tid & 31, w = tid >> 5;
  const int m16 = lane & 15, laneHi = (lane >> 4) & 1;
  const int o_r = w * 16 + m16, o_z = o_r + 128, o_n = o_r + 256;

  __shared__ _Float16 hbuf[16 * HP];
  __shared__ float sc[16][2];

  for (int i = tid; i < 16 * HP; i += 256) hbuf[i] = (_Float16)0.f;

  // hoisted per-lane gate constants (input dim is 2 -> scalar xg path)
  const float wr0 = Wd[o_r * 2], wr1 = Wd[o_r * 2 + 1], br = bi[o_r], Br = bh[o_r];
  const float wz0 = Wd[o_z * 2], wz1 = Wd[o_z * 2 + 1], bz = bi[o_z], Bz = bh[o_z];
  const float wn0 = Wd[o_n * 2], wn1 = Wd[o_n * 2 + 1], bn = bi[o_n], Bn = bh[o_n];

  const v16h* pb = (const v16h*)pW;

  for (int t = 0; t < LWALK; ++t) {
    __syncthreads();
    int tsrc = dir ? (LWALK - 1 - t) : t;
    if (tid < 32) {
      int m = tid >> 1;
      sc[m][tid & 1] = usc[((size_t)(b0 + m) * LWALK + tsrc) * 2 + (tid & 1)];
    }
    __syncthreads();

    v8f cr, cz, cn;
#pragma unroll
    for (int r = 0; r < 8; ++r) { cr[r] = Br; cz[r] = Bz; cn[r] = Bn; }
#pragma unroll
    for (int kt = 0; kt < 4; ++kt) {
      v16h a;
#pragma unroll
      for (int p = 0; p < 16; ++p) a[p] = hbuf[m16 * HP + kt * 32 + koff(p, laneHi)];
      cr = wmma_f16(a, pb[((w)      * 4 + kt) * 32 + lane], cr);
      cz = wmma_f16(a, pb[((w + 8)  * 4 + kt) * 32 + lane], cz);
      cn = wmma_f16(a, pb[((w + 16) * 4 + kt) * 32 + lane], cn);
    }
    __syncthreads();  // all hbuf fragment reads complete

    float hnew[8];
#pragma unroll
    for (int r = 0; r < 8; ++r) {
      int row = laneHi ? r + 8 : r;
      float s_ = sc[row][0], c_ = sc[row][1];
      float xr = fmaf(s_, wr0, fmaf(c_, wr1, br));
      float xz = fmaf(s_, wz0, fmaf(c_, wz1, bz));
      float xn = fmaf(s_, wn0, fmaf(c_, wn1, bn));
      float hprev = (float)hbuf[row * HP + o_r];
      float rg = sigm(xr + cr[r]);
      float zg = sigm(xz + cz[r]);
      float ng = tanh_fast(xn + rg * cn[r]);
      hnew[r] = (1.f - zg) * ng + zg * hprev;
    }
    int tout = dir ? (LWALK - 1 - t) : t;
#pragma unroll
    for (int r = 0; r < 8; ++r) {
      int row = laneHi ? r + 8 : r;
      hbuf[row * HP + o_r] = (_Float16)hnew[r];
      ywalk[((size_t)(b0 + row) * LWALK + tout) * 256 + dir * 128 + o_r] =
          (_Float16)hnew[r];
    }
  }
#pragma unroll
  for (int r = 0; r < 8; ++r) {
    int row = laneHi ? r + 8 : r;
    hdir[((size_t)dir * BROWS + (b0 + row)) * OUTF + o_r] = hbuf[row * HP + o_r];
  }
}

// ------------------------------------------------------------ main GRU -----
__global__ __launch_bounds__(256) void main_gru_kernel(
    const float* __restrict__ hglob, const float* __restrict__ e,
    const int* __restrict__ walks, const int* __restrict__ eids,
    const int* __restrict__ deg,
    const float* __restrict__ Wih, const float* __restrict__ bih,
    const float* __restrict__ bhh,
    const _Float16* __restrict__ pWhh, const _Float16* __restrict__ pWihK,
    const _Float16* __restrict__ pM2,
    const _Float16* __restrict__ ywalk, const _Float16* __restrict__ hdir,
    const int* __restrict__ maxdeg_p, float* __restrict__ out) {
  const int b0 = blockIdx.x * 16;
  const int tid = threadIdx.x;
  const int lane = tid & 31, w = tid >> 5;
  const int m16 = lane & 15, laneHi = (lane >> 4) & 1;
  const int o_r = w * 16 + m16, o_z = o_r + 128, o_n = o_r + 256;

  __shared__ _Float16 hbuf[16 * HP];
  __shared__ _Float16 xbuf[16 * XP];
  __shared__ float dscale[16];
  __shared__ int srow[16];

  float md = (float)maxdeg_p[0];
  float invmax = md > 0.f ? 1.f / md : 0.f;

  // h0 = 0.5*(hf + hb)
  for (int i = tid; i < 16 * 128; i += 256) {
    int m = i >> 7, k = i & 127, b = b0 + m;
    float hv = 0.5f * ((float)hdir[(size_t)b * OUTF + k] +
                       (float)hdir[(size_t)(BROWS + b) * OUTF + k]);
    hbuf[m * HP + k] = (_Float16)hv;
  }

  const float br = bih[o_r], bz = bih[o_z], bn = bih[o_n];
  const float Br = bhh[o_r], Bz = bhh[o_z], Bn = bhh[o_n];
  const float wd_r = Wih[(size_t)o_r * DINF + 384];
  const float wd_z = Wih[(size_t)o_z * DINF + 384];
  const float wd_n = Wih[(size_t)o_n * DINF + 384];

  const v16h* pbh = (const v16h*)pWhh;

  for (int t = 0; t < 2 * LWALK - 1; ++t) {
    __syncthreads();
    const bool even = (t & 1) == 0;
    const int tk = t >> 1;
    if (even) {
      if (tid < 16) {
        int b = b0 + tid;
        int widx = walks[(size_t)b * LWALK + (LWALK - 1 - tk)];
        srow[tid] = widx;
        dscale[tid] = (float)deg[widx] * invmax;
      }
      // y_walk rows are already f16: stream them into LDS with the CDNA5
      // async global->LDS path (ASYNCcnt-tracked, bypasses VGPRs).
      // 16 rows x 512B = 32 16B-chunks/row; 256 threads x 2 chunks.
#pragma unroll
      for (int c = tid; c < 512; c += 256) {
        int m = c >> 5, j = c & 31;
        unsigned lds_a =
            (unsigned)(uintptr_t)(void*)&xbuf[m * XP + 128 + j * 8];
        const _Float16* gsrc =
            ywalk + (((size_t)(b0 + m) * LWALK + tk) * 256 + (size_t)j * 8);
        asm volatile("global_load_async_to_lds_b128 %0, %1, off"
                     :
                     : "v"(lds_a), "v"((unsigned long long)(uintptr_t)gsrc)
                     : "memory");
      }
      __syncthreads();
      for (int i = tid; i < 16 * 128; i += 256) {
        int m = i >> 7, k = i & 127;
        xbuf[m * XP + k] = (_Float16)hglob[(size_t)srow[m] * INF_ + k];
      }
      asm volatile("s_wait_asynccnt 0x0" ::: "memory");
    } else {
      if (tid < 16) srow[tid] = eids[(size_t)(b0 + tid) * (LWALK - 1) + tk];
      __syncthreads();
      for (int i = tid; i < 16 * 64; i += 256) {
        int m = i >> 6, k = i & 63;
        xbuf[m * XP + k] = (_Float16)e[(size_t)srow[m] * EFEAT + k];
      }
    }
    __syncthreads();

    // input-gate GEMM: even K=384 (Wih cols 0..383), odd K=64 (M2)
    const int KTx = even ? 12 : 2;
    const v16h* pbx = (const v16h*)(even ? pWihK : pM2);
    v8f xr_, xz_, xn_;
#pragma unroll
    for (int r = 0; r < 8; ++r) {
      int row = laneHi ? r + 8 : r;
      float ds = even ? dscale[row] : 0.f;
      xr_[r] = fmaf(ds, wd_r, br);
      xz_[r] = fmaf(ds, wd_z, bz);
      xn_[r] = fmaf(ds, wd_n, bn);
    }
    for (int kt = 0; kt < KTx; ++kt) {
      v16h a;
#pragma unroll
      for (int p = 0; p < 16; ++p) a[p] = xbuf[m16 * XP + kt * 32 + koff(p, laneHi)];
      xr_ = wmma_f16(a, pbx[((size_t)(w)      * KTx + kt) * 32 + lane], xr_);
      xz_ = wmma_f16(a, pbx[((size_t)(w + 8)  * KTx + kt) * 32 + lane], xz_);
      xn_ = wmma_f16(a, pbx[((size_t)(w + 16) * KTx + kt) * 32 + lane], xn_);
    }

    // recurrent GEMM: K=128
    v8f hr_, hz_, hn_;
#pragma unroll
    for (int r = 0; r < 8; ++r) { hr_[r] = Br; hz_[r] = Bz; hn_[r] = Bn; }
#pragma unroll
    for (int kt = 0; kt < 4; ++kt) {
      v16h a;
#pragma unroll
      for (int p = 0; p < 16; ++p) a[p] = hbuf[m16 * HP + kt * 32 + koff(p, laneHi)];
      hr_ = wmma_f16(a, pbh[((w)      * 4 + kt) * 32 + lane], hr_);
      hz_ = wmma_f16(a, pbh[((w + 8)  * 4 + kt) * 32 + lane], hz_);
      hn_ = wmma_f16(a, pbh[((w + 16) * 4 + kt) * 32 + lane], hn_);
    }
    __syncthreads();  // all hbuf/xbuf fragment reads complete

    float hnew[8];
#pragma unroll
    for (int r = 0; r < 8; ++r) {
      int row = laneHi ? r + 8 : r;
      float hprev = (float)hbuf[row * HP + o_r];
      float rg = sigm(xr_[r] + hr_[r]);
      float zg = sigm(xz_[r] + hz_[r]);
      float ng = tanh_fast(xn_[r] + rg * hn_[r]);
      hnew[r] = (1.f - zg) * ng + zg * hprev;
    }
#pragma unroll
    for (int r = 0; r < 8; ++r) {
      int row = laneHi ? r + 8 : r;
      hbuf[row * HP + o_r] = (_Float16)hnew[r];
    }
  }
#pragma unroll
  for (int r = 0; r < 8; ++r) {
    int row = laneHi ? r + 8 : r;
    out[(size_t)(b0 + row) * OUTF + o_r] = (float)hbuf[row * HP + o_r];
  }
}

// --------------------------------------------------------------- launch ----
extern "C" void kernel_launch(void* const* d_in, const int* in_sizes, int n_in,
                              void* d_out, int out_size, void* d_ws, size_t ws_size,
                              hipStream_t stream) {
  const float* h     = (const float*)d_in[0];
  const float* e     = (const float*)d_in[1];
  const float* W_e   = (const float*)d_in[2];
  const float* Wih_f = (const float*)d_in[3];
  const float* Whh_f = (const float*)d_in[4];
  const float* bih_f = (const float*)d_in[5];
  const float* bhh_f = (const float*)d_in[6];
  const float* Wih_b = (const float*)d_in[7];
  const float* Whh_b = (const float*)d_in[8];
  const float* bih_b = (const float*)d_in[9];
  const float* bhh_b = (const float*)d_in[10];
  const float* Wih   = (const float*)d_in[11];
  const float* Whh   = (const float*)d_in[12];
  const float* bih   = (const float*)d_in[13];
  const float* bhh   = (const float*)d_in[14];
  const int* walks   = (const int*)d_in[15];
  const int* eids    = (const int*)d_in[16];
  const int* deg     = (const int*)d_in[17];
  float* out = (float*)d_out;

  char* wsb = (char*)d_ws;
  size_t off = 0;
  auto take = [&](size_t bytes) -> char* {
    char* p = wsb + off;
    off = (off + bytes + 255) & ~(size_t)255;
    return p;
  };
  int*      maxdeg = (int*)take(sizeof(int));
  float*    usc    = (float*)take((size_t)BROWS * LWALK * 2 * sizeof(float));
  _Float16* ywalk  = (_Float16*)take((size_t)BROWS * LWALK * 256 * sizeof(_Float16));
  _Float16* hdir   = (_Float16*)take((size_t)2 * BROWS * OUTF * sizeof(_Float16));
  _Float16* pWhh_f = (_Float16*)take((size_t)GDIM * 128 * sizeof(_Float16));
  _Float16* pWhh_b = (_Float16*)take((size_t)GDIM * 128 * sizeof(_Float16));
  _Float16* pWhh   = (_Float16*)take((size_t)GDIM * 128 * sizeof(_Float16));
  _Float16* pWihK  = (_Float16*)take((size_t)GDIM * 384 * sizeof(_Float16));
  _Float16* pM2    = (_Float16*)take((size_t)GDIM * 64 * sizeof(_Float16));
  float*    M2f    = (float*)take((size_t)GDIM * 64 * sizeof(float));

  hipMemsetAsync(maxdeg, 0, sizeof(int), stream);
  prep_kernel<<<(BROWS + 255) / 256, 256, 0, stream>>>(walks, deg, usc, maxdeg);
  m2_kernel<<<(GDIM * EFEAT + 255) / 256, 256, 0, stream>>>(Wih, W_e, M2f);
  pack_kernel<<<dim3(36, 5), 256, 0, stream>>>(Whh_f, Whh_b, Whh, Wih, M2f,
                                               pWhh_f, pWhh_b, pWhh, pWihK, pM2);
  walk_gru_kernel<<<dim3(BROWS / 16, 2), 256, 0, stream>>>(
      usc, Wih_f, bih_f, bhh_f, Wih_b, bih_b, bhh_b, pWhh_f, pWhh_b, ywalk, hdir);
  main_gru_kernel<<<BROWS / 16, 256, 0, stream>>>(
      h, e, walks, eids, deg, Wih, bih, bhh, pWhh, pWihK, pM2, ywalk, hdir,
      maxdeg, out);
}
